// TripletIMP_13924283974459
// MI455X (gfx1250) — matrix operand order repaired and checked by reading, not compiled, not run on hardware
//
#include <hip/hip_runtime.h>
#include <hip/hip_bf16.h>
#include <math.h>

#define DIM 128
#define NN  50000
#define NE  800000

typedef __attribute__((ext_vector_type(16))) _Float16 v16h;
typedef __attribute__((ext_vector_type(8)))  _Float16 v8h;
typedef __attribute__((ext_vector_type(8)))  float    v8f;
typedef __attribute__((ext_vector_type(4)))  float    v4f;
typedef __attribute__((ext_vector_type(4)))  unsigned int u32x4;
typedef __attribute__((ext_vector_type(8)))  int      i32x8;
typedef __attribute__((ext_vector_type(4)))  int      i32x4;

#if defined(__HIP_DEVICE_COMPILE__) && __has_builtin(__builtin_amdgcn_tensor_load_to_lds) && \
    __has_builtin(__builtin_amdgcn_s_wait_tensorcnt)
#define HAVE_TDM 1
#else
#define HAVE_TDM 0
#endif

__device__ __forceinline__ float fast_rcp(float x) {
#if defined(__HIP_DEVICE_COMPILE__) && __has_builtin(__builtin_amdgcn_rcpf)
    return __builtin_amdgcn_rcpf(x);
#else
    return 1.0f / x;
#endif
}
__device__ __forceinline__ float fast_sigmoid(float x) { return fast_rcp(1.0f + __expf(-x)); }
__device__ __forceinline__ float fast_tanh(float x)    { return 1.0f - 2.0f * fast_rcp(1.0f + __expf(2.0f * x)); }

__device__ __forceinline__ v8f wmma_f16(v16h a, v16h b, v8f c) {
    // (neg_a, A, neg_b, B, c_mod, C, reuse_a, reuse_b)
    return __builtin_amdgcn_wmma_f32_16x16x32_f16(false, a, false, b, (short)0, c, false, false);
}

#if HAVE_TDM
// TDM: DMA a contiguous run of `nelem` f16 elements (nelem < 65536) from global to LDS.
// D# built per CDNA5 ISA: group0 = {count=1, lds_addr, global_addr, type=2},
// group1 = {data_size=2B, tensor_dim0=nelem, tile_dim0=nelem, stride=nelem}.
__device__ __forceinline__ void tdm_load_f16(const void* gptr, unsigned lds_byte_off, unsigned nelem) {
    unsigned long long ga = (unsigned long long)(size_t)gptr;
    u32x4 g0;
    g0[0] = 1u;                                               // count=1 (user descriptor)
    g0[1] = lds_byte_off;                                     // lds_addr (bytes)
    g0[2] = (unsigned)(ga & 0xFFFFFFFFu);                     // global_addr[31:0]
    g0[3] = (unsigned)((ga >> 32) & 0x01FFFFFFu) | (2u << 30);// global_addr[56:32] | type=2
    i32x8 g1;
    g1[0] = (int)(1u << 16);                                  // data_size=1 (2 bytes)
    g1[1] = (int)((nelem & 0xFFFFu) << 16);                   // tensor_dim0[15:0] @ bits63:48
    g1[2] = (int)(((nelem >> 16) & 0xFFFFu) | (1u << 16));    // tensor_dim0[31:16], tensor_dim1=1
    g1[3] = (int)((nelem & 0xFFFFu) << 16);                   // tile_dim0 @ bits127:112
    g1[4] = 0;                                                // tile_dim1=0, tile_dim2=0
    g1[5] = (int)nelem;                                       // tensor_dim0_stride[31:0]
    g1[6] = 0;
    g1[7] = 0;
    i32x4 z4 = {0, 0, 0, 0};
#if defined(__clang_major__) && (__clang_major__ >= 23)
    i32x8 z8 = {0, 0, 0, 0, 0, 0, 0, 0};
    __builtin_amdgcn_tensor_load_to_lds(g0, g1, z4, z4, z8, 0);
#else
    __builtin_amdgcn_tensor_load_to_lds(g0, g1, z4, z4, 0);
#endif
}
#endif

// ---------------- utility kernels ----------------

__global__ void k_zero(float* __restrict__ p, int n) {
    int i = blockIdx.x * blockDim.x + threadIdx.x;
    if (i < n) p[i] = 0.0f;
}

// f32 -> f16 convert; if cnt != null, divides element by max(cnt[row],1) (row = i/128)
__global__ void k_f32_to_f16(const float* __restrict__ src, _Float16* __restrict__ dst,
                             const float* __restrict__ cnt, int n) {
    int i = blockIdx.x * blockDim.x + threadIdx.x;
    if (i >= n) return;
    float v = src[i];
    if (cnt) v *= fast_rcp(fmaxf(cnt[i >> 7], 1.0f));
    dst[i] = (_Float16)v;
}

__global__ void k_count(const int* __restrict__ eidx, float* __restrict__ cnt, int nE) {
    int e = blockIdx.x * blockDim.x + threadIdx.x;
    if (e < nE) atomicAdd(&cnt[eidx[nE + e]], 1.0f);
}

// ---------------- edge gate/message kernel: one wave per edge (grid-strided) ----------------

__global__ __launch_bounds__(256) void k_edge(
    const float* __restrict__ x, const float* __restrict__ ef,
    const int* __restrict__ eidx,
    const float* __restrict__ Wsn, const float* __restrict__ bsn,
    const float* __restrict__ Won, const float* __restrict__ bon,
    const float* __restrict__ Wse, const float* __restrict__ bse,
    const float* __restrict__ Woe, const float* __restrict__ boe,
    float* __restrict__ edge_msg, float* __restrict__ node_acc, int nE)
{
    __shared__ float wg[4][2][DIM];     // gate weight vectors
    for (int i = threadIdx.x; i < 2 * DIM; i += blockDim.x) {
        wg[0][i >> 7][i & 127] = Wsn[i];
        wg[1][i >> 7][i & 127] = Won[i];
        wg[2][i >> 7][i & 127] = Wse[i];
        wg[3][i >> 7][i & 127] = Woe[i];
    }
    __syncthreads();

    const int lane   = threadIdx.x & 31;
    const int wavesB = blockDim.x >> 5;
    const int stride = gridDim.x * wavesB;
    const float b0 = bsn[0], b1 = bon[0], b2 = bse[0], b3 = boe[0];

    for (int e = blockIdx.x * wavesB + (threadIdx.x >> 5); e < nE; e += stride) {
        if (e + stride < nE) {
            __builtin_prefetch(ef + (size_t)(e + stride) * DIM + lane * 4, 0, 0);
            __builtin_prefetch(eidx + e + stride, 0, 0);
        }
        const int s = eidx[e];       // src
        const int d = eidx[nE + e];  // dst
        v4f xi = *(const v4f*)(x  + (size_t)d * DIM + lane * 4);
        v4f xj = *(const v4f*)(x  + (size_t)s * DIM + lane * 4);
        v4f e4 = *(const v4f*)(ef + (size_t)e * DIM + lane * 4);

        float p_s = 0.f, p_o = 0.f, p_es = 0.f, p_eo = 0.f;
        #pragma unroll
        for (int u = 0; u < 4; ++u) {
            const int c = lane * 4 + u;
            p_s  += xi[u] * wg[0][0][c] + e4[u] * wg[0][1][c];
            p_o  += xj[u] * wg[1][0][c] + e4[u] * wg[1][1][c];
            p_es += xi[u] * wg[2][0][c] + e4[u] * wg[2][1][c];
            p_eo += xj[u] * wg[3][0][c] + e4[u] * wg[3][1][c];
        }
        #pragma unroll
        for (int off = 16; off > 0; off >>= 1) {
            p_s  += __shfl_xor(p_s,  off, 32);
            p_o  += __shfl_xor(p_o,  off, 32);
            p_es += __shfl_xor(p_es, off, 32);
            p_eo += __shfl_xor(p_eo, off, 32);
        }
        const float g_s  = fast_sigmoid(p_s  + b0);
        const float g_o  = fast_sigmoid(p_o  + b1);
        const float ge_s = fast_sigmoid(p_es + b2);
        const float ge_o = fast_sigmoid(p_eo + b3);

        float* nacc = node_acc + (size_t)d * DIM;
        v4f em;
        #pragma unroll
        for (int u = 0; u < 4; ++u) {
            const int c = lane * 4 + u;
            atomicAdd(&nacc[c], (g_s + g_o) * e4[u]);   // segment_sum scatter
            em[u] = ge_s * xi[u] + ge_o * xj[u];
        }
        *(v4f*)(edge_msg + (size_t)e * DIM + lane * 4) = em;
    }
}

// ---------------- fused GRU cell: one wave per 16-row strip, WMMA over K=128 ----------------
// Weights live in LDS (staged via TDM); B tile reads are ds_load_b128 so the WMMA pipe never
// waits on global latency. Xh/Hh: f16 row-major (rows x 128); W*: f16 row-major (384 x 128).

__device__ __forceinline__ v16h loadB_lds(const _Float16* __restrict__ W, int colbase, int kt, int lane) {
    const int n = lane & 15, kh = lane >> 4;
    const _Float16* p = W + (size_t)(colbase + n) * DIM + kt * 32 + kh * 8;
    v8h lo = *(const v8h*)(p);
    v8h hi = *(const v8h*)(p + 16);
    v16h r;
    #pragma unroll
    for (int i = 0; i < 8; ++i) { r[i] = lo[i]; r[8 + i] = hi[i]; }
    return r;
}

#define NWELEM (3 * DIM * DIM)   // 49152 f16 elements per weight matrix

template <bool HAS_H>
__global__ __launch_bounds__(256) void k_gru(
    const _Float16* __restrict__ Xh, const _Float16* __restrict__ Hh,
    const float* __restrict__ H32,
    const _Float16* __restrict__ Wih, const _Float16* __restrict__ Whh,
    const float* __restrict__ bih, const float* __restrict__ bhh,
    float* __restrict__ out, int nRows)
{
    extern __shared__ _Float16 smem[];          // Wih [+ Whh]
    _Float16* sWih = smem;
    _Float16* sWhh = smem + NWELEM;

#if HAVE_TDM
    if ((threadIdx.x >> 5) == 0) {              // wave 0 issues the tensor DMAs
        tdm_load_f16(Wih, (unsigned)(size_t)sWih, NWELEM);
        if (HAS_H) tdm_load_f16(Whh, (unsigned)(size_t)sWhh, NWELEM);
        __builtin_amdgcn_s_wait_tensorcnt(0);
    }
#else
    for (int i = threadIdx.x; i < NWELEM; i += blockDim.x) sWih[i] = Wih[i];
    if (HAS_H) for (int i = threadIdx.x; i < NWELEM; i += blockDim.x) sWhh[i] = Whh[i];
#endif
    __syncthreads();

    const int lane = threadIdx.x & 31;
    const int wave = blockIdx.x * (blockDim.x >> 5) + (threadIdx.x >> 5);
    const int row0 = wave * 16;
    if (row0 >= nRows) return;                  // wave-uniform: EXEC all-ones for WMMA

    const int m = lane & 15, kh = lane >> 4;

    // A fragments (16x32 f16 per k-tile): lane = row m; VGPR0-3 => K=kh*8+0..7, VGPR4-7 => +16
    v16h ax[4], ah[4];
    {
        const _Float16* xr = Xh + (size_t)(row0 + m) * DIM;
        #pragma unroll
        for (int kt = 0; kt < 4; ++kt) {
            v8h lo = *(const v8h*)(xr + kt * 32 + kh * 8);
            v8h hi = *(const v8h*)(xr + kt * 32 + 16 + kh * 8);
            #pragma unroll
            for (int i = 0; i < 8; ++i) { ax[kt][i] = lo[i]; ax[kt][8 + i] = hi[i]; }
        }
        if (HAS_H) {
            const _Float16* hr = Hh + (size_t)(row0 + m) * DIM;
            #pragma unroll
            for (int kt = 0; kt < 4; ++kt) {
                v8h lo = *(const v8h*)(hr + kt * 32 + kh * 8);
                v8h hi = *(const v8h*)(hr + kt * 32 + 16 + kh * 8);
                #pragma unroll
                for (int i = 0; i < 8; ++i) { ah[kt][i] = lo[i]; ah[kt][8 + i] = hi[i]; }
            }
        }
    }

    for (int c = 0; c < 8; ++c) {               // 8 column chunks of 16 => 128 output cols
        v8f a_ir = {}, a_iz = {}, a_in = {}, a_hr = {}, a_hz = {}, a_hn = {};
        #pragma unroll
        for (int kt = 0; kt < 4; ++kt) {
            a_ir = wmma_f16(ax[kt], loadB_lds(sWih,        c * 16, kt, lane), a_ir);
            a_iz = wmma_f16(ax[kt], loadB_lds(sWih, 128 +  c * 16, kt, lane), a_iz);
            a_in = wmma_f16(ax[kt], loadB_lds(sWih, 256 +  c * 16, kt, lane), a_in);
            if (HAS_H) {
                a_hr = wmma_f16(ah[kt], loadB_lds(sWhh,        c * 16, kt, lane), a_hr);
                a_hz = wmma_f16(ah[kt], loadB_lds(sWhh, 128 +  c * 16, kt, lane), a_hz);
                a_hn = wmma_f16(ah[kt], loadB_lds(sWhh, 256 +  c * 16, kt, lane), a_hn);
            }
        }
        const int col = c * 16 + m;             // C/D: lane -> column (l&15)
        #pragma unroll
        for (int v = 0; v < 8; ++v) {           // C/D: element v -> row v + 8*(l>>4)
            const int row = row0 + v + 8 * kh;
            const float ir  = a_ir[v] + bih[col];
            const float iz  = a_iz[v] + bih[128 + col];
            const float in_ = a_in[v] + bih[256 + col];
            float hr = bhh[col], hz = bhh[128 + col], hn = bhh[256 + col];
            if (HAS_H) { hr += a_hr[v]; hz += a_hz[v]; hn += a_hn[v]; }
            const float r  = fast_sigmoid(ir + hr);
            const float z  = fast_sigmoid(iz + hz);
            const float nn = fast_tanh(in_ + r * hn);
            const float h  = HAS_H ? H32[(size_t)row * DIM + col] : 0.0f;
            out[(size_t)row * DIM + col] = (1.0f - z) * nn + z * h;
        }
    }
}

// ---------------- host orchestration ----------------

extern "C" void kernel_launch(void* const* d_in, const int* in_sizes, int n_in,
                              void* d_out, int out_size, void* d_ws, size_t ws_size,
                              hipStream_t stream) {
    const float* x_in  = (const float*)d_in[0];
    const float* ef_in = (const float*)d_in[1];
    const int*   eidx  = (const int*)d_in[2];
    const float* Wih_n = (const float*)d_in[3];
    const float* Whh_n = (const float*)d_in[4];
    const float* bih_n = (const float*)d_in[5];
    const float* bhh_n = (const float*)d_in[6];
    const float* Wih_e = (const float*)d_in[7];
    const float* Whh_e = (const float*)d_in[8];
    const float* bih_e = (const float*)d_in[9];
    const float* bhh_e = (const float*)d_in[10];
    const float* W_sn = (const float*)d_in[11]; const float* b_sn = (const float*)d_in[12];
    const float* W_on = (const float*)d_in[13]; const float* b_on = (const float*)d_in[14];
    const float* W_se = (const float*)d_in[15]; const float* b_se = (const float*)d_in[16];
    const float* W_oe = (const float*)d_in[17]; const float* b_oe = (const float*)d_in[18];

    float* out_x  = (float*)d_out;                       // 50000 x 128
    float* out_ef = out_x + (size_t)NN * DIM;            // 800000 x 128

    char* ws = (char*)d_ws;
    size_t off = 0;
    auto alloc = [&](size_t bytes) { void* p = ws + off; off = (off + bytes + 255) & ~(size_t)255; return p; };

    _Float16* Wn_ih_h = (_Float16*)alloc((size_t)NWELEM * 2);
    _Float16* Wn_hh_h = (_Float16*)alloc((size_t)NWELEM * 2);
    _Float16* We_ih_h = (_Float16*)alloc((size_t)NWELEM * 2);
    _Float16* We_hh_h = (_Float16*)alloc((size_t)NWELEM * 2);
    float*    cnt      = (float*)alloc((size_t)NN * 4);
    float*    node_acc = (float*)alloc((size_t)NN * DIM * 4);
    float*    edge_msg = (float*)alloc((size_t)NE * DIM * 4);
    _Float16* Xh       = (_Float16*)alloc((size_t)NE * DIM * 2);
    _Float16* Hh       = (_Float16*)alloc((size_t)NE * DIM * 2);

    const int TB = 256;
    const unsigned LDS1 = (unsigned)NWELEM * 2;          // Wih only (initial GRU, h=0)
    const unsigned LDS2 = (unsigned)NWELEM * 4;          // Wih + Whh
    auto blk = [](long long n, int tb) { return (unsigned)((n + tb - 1) / tb); };

    k_f32_to_f16<<<blk(NWELEM, TB), TB, 0, stream>>>(Wih_n, Wn_ih_h, nullptr, NWELEM);
    k_f32_to_f16<<<blk(NWELEM, TB), TB, 0, stream>>>(Whh_n, Wn_hh_h, nullptr, NWELEM);
    k_f32_to_f16<<<blk(NWELEM, TB), TB, 0, stream>>>(Wih_e, We_ih_h, nullptr, NWELEM);
    k_f32_to_f16<<<blk(NWELEM, TB), TB, 0, stream>>>(Whh_e, We_hh_h, nullptr, NWELEM);

    k_zero<<<blk(NN, TB), TB, 0, stream>>>(cnt, NN);
    k_count<<<blk(NE, TB), TB, 0, stream>>>(eidx, cnt, NE);

    // initial GRUs (h = 0)
    const long long nNodeE = (long long)NN * DIM, nEdgeE = (long long)NE * DIM;
    k_f32_to_f16<<<blk(nNodeE, TB), TB, 0, stream>>>(x_in, Xh, nullptr, (int)nNodeE);
    k_gru<false><<<blk(NN / 16, 8), TB, LDS1, stream>>>(Xh, nullptr, nullptr, Wn_ih_h, Wn_hh_h,
                                                        bih_n, bhh_n, out_x, NN);
    k_f32_to_f16<<<blk(nEdgeE, TB), TB, 0, stream>>>(ef_in, Xh, nullptr, (int)nEdgeE);
    k_gru<false><<<blk(NE / 16, 8), TB, LDS1, stream>>>(Xh, nullptr, nullptr, We_ih_h, We_hh_h,
                                                        bih_e, bhh_e, out_ef, NE);

    for (int layer = 0; layer < 2; ++layer) {
        k_zero<<<blk(nNodeE, TB), TB, 0, stream>>>(node_acc, (int)nNodeE);
        k_edge<<<8192, TB, 0, stream>>>(out_x, out_ef, eidx,
                                        W_sn, b_sn, W_on, b_on, W_se, b_se, W_oe, b_oe,
                                        edge_msg, node_acc, NE);
        // node GRU: x = GRU(node_acc / max(cnt,1), x)
        k_f32_to_f16<<<blk(nNodeE, TB), TB, 0, stream>>>(node_acc, Xh, cnt, (int)nNodeE);
        k_f32_to_f16<<<blk(nNodeE, TB), TB, 0, stream>>>(out_x, Hh, nullptr, (int)nNodeE);
        k_gru<true><<<blk(NN / 16, 8), TB, LDS2, stream>>>(Xh, Hh, out_x, Wn_ih_h, Wn_hh_h,
                                                           bih_n, bhh_n, out_x, NN);
        // edge GRU: ef = GRU(edge_msg, ef)
        k_f32_to_f16<<<blk(nEdgeE, TB), TB, 0, stream>>>(edge_msg, Xh, nullptr, (int)nEdgeE);
        k_f32_to_f16<<<blk(nEdgeE, TB), TB, 0, stream>>>(out_ef, Hh, nullptr, (int)nEdgeE);
        k_gru<true><<<blk(NE / 16, 8), TB, LDS2, stream>>>(Xh, Hh, out_ef, We_ih_h, We_hh_h,
                                                           bih_e, bhh_e, out_ef, NE);
    }
    (void)in_sizes; (void)n_in; (void)out_size; (void)ws_size;
}
